// DoubleTransformer_28106265985228
// MI455X (gfx1250) — compile-verified, matrix operand order checked by reading, not run on hardware
//
#include <hip/hip_runtime.h>
#include <hip/hip_bf16.h>
#include <math.h>

// ---------------------------------------------------------------------------
// DoubleTransformer for MI455X (gfx1250, wave32, WMMA).
// All GEMMs -> v_wmma_f32_16x16x32_bf16 (bf16 in, f32 accumulate).
// Attention: two-pass (scores f32 -> softmax -> probs bf16 -> P@V), relying on
// the 192MB L2 to keep the 56-113MB score matrices cache-resident.
// GEMM K-loop uses double-buffered LDS (one barrier per chunk) plus
// global_prefetch_b8 hints two chunks ahead.
// Workspace requirement: ~560 MB.
// ---------------------------------------------------------------------------

typedef __attribute__((ext_vector_type(16))) __bf16 v16bf;
typedef __attribute__((ext_vector_type(8)))  float  v8f;

#define TB   8
#define TV   48
#define TL   95
#define TD   512
#define TFF  2048
#define TNC  3
#define TE   96        // L+1
#define NTOK (TB*TV*TE)  // 36864 tokens in both stages

// ---- tiled WMMA GEMM: C[M,N] = act(alpha * A @ B^T + bias) ----------------
// A: bf16, row-major M x K with leading dim lda (elements).
// B: bf16. If B_KN==false: B stored N x K row-major (ldb) -> C = A @ B^T.
//          If B_KN==true : B stored K x N row-major (ldb) -> C = A @ B.
// Per-z (blockIdx.z) base offsets allow strided sub-matrices (per-slot attn).
// Requires: M%64==0, N%128==0, K%32==0 (true for every call in this model).
#define BM 64
#define BN 128
#define KC 32
#define LDSP 40   // padded LDS row stride (elements); 80B rows, conflict-free

template<typename OT, bool B_KN>
__global__ __launch_bounds__(128)
void gemm_wmma(const __bf16* __restrict__ A, const __bf16* __restrict__ B,
               OT* __restrict__ C, const float* __restrict__ bias,
               int M, int N, int K, long lda, long ldb, long ldc,
               long aOff0, long aStrideZ, long bOff0, long bStrideZ,
               long cOff0, long cStrideZ, float alpha, int relu)
{
    __shared__ __align__(16) __bf16 As[2][BM * LDSP];
    __shared__ __align__(16) __bf16 Bs[2][BN * LDSP];

    const int z = blockIdx.z;
    A += aOff0 + (long)z * aStrideZ;
    B += bOff0 + (long)z * bStrideZ;
    C += cOff0 + (long)z * cStrideZ;

    const int m0  = blockIdx.y * BM;
    const int n0  = blockIdx.x * BN;
    const int tid = threadIdx.x;
    const int lane = tid & 31;
    const int wid  = tid >> 5;
    const int wm = (wid & 1) * 32;   // wave's row offset in block tile
    const int wn = (wid >> 1) * 64;  // wave's col offset in block tile
    const int h  = lane >> 4;        // lane half (ISA bf16 fragment layout)
    const int ml = lane & 15;

    v8f acc[2][4];
#pragma unroll
    for (int mi = 0; mi < 2; ++mi)
#pragma unroll
        for (int ni = 0; ni < 4; ++ni)
#pragma unroll
            for (int r = 0; r < 8; ++r) acc[mi][ni][r] = 0.0f;

    // ---- global -> LDS tile stage (double-buffered) ----
    auto load_tiles = [&](int k0, int buf) {
#pragma unroll
        for (int i = 0; i < 2; ++i) {
            int li = tid + i * 128;          // 0..255
            int r  = li >> 2;                // row 0..63
            int cb = (li & 3) * 8;           // 0,8,16,24
            *(uint4*)(&As[buf][r * LDSP + cb]) =
                *(const uint4*)(A + (long)(m0 + r) * lda + k0 + cb);
        }
        if (!B_KN) {
            int n = tid;                     // 0..127 (BN rows)
            const uint4* gp = (const uint4*)(B + (long)(n0 + n) * ldb + k0);
            uint4* sp = (uint4*)(&Bs[buf][n * LDSP]);
            sp[0] = gp[0]; sp[1] = gp[1]; sp[2] = gp[2]; sp[3] = gp[3];
        } else {
            // B is KxN: transpose while staging
#pragma unroll 4
            for (int i = 0; i < 32; ++i) {
                int e  = tid * 32 + i;       // 0..4095
                int n  = e >> 5;
                int kk = e & 31;
                Bs[buf][n * LDSP + kk] = B[(long)(k0 + kk) * ldb + (n0 + n)];
            }
        }
    };

    const int nk = K / KC;
    load_tiles(0, 0);

    for (int kc = 0; kc < nk; ++kc) {
        __syncthreads();
        if (kc + 1 < nk) {
            if (kc + 2 < nk) {
                // pre-warm L2 two chunks ahead (global_prefetch_b8)
                int kp = (kc + 2) * KC;
                __builtin_prefetch(A + (long)(m0 + (tid >> 1)) * lda + kp, 0, 1);
                if (!B_KN)
                    __builtin_prefetch(B + (long)(n0 + tid) * ldb + kp, 0, 1);
            }
            load_tiles((kc + 1) * KC, (kc + 1) & 1);
        }
        const int buf = kc & 1;

        // --- fragments per CDNA5 16-bit layout:
        //     lane half h holds K in {8h..8h+7} U {16+8h..16+8h+7}, paired.
        v16bf afrag[2];
#pragma unroll
        for (int mi = 0; mi < 2; ++mi) {
            const __bf16* ap = &As[buf][(wm + mi * 16 + ml) * LDSP];
#pragma unroll
            for (int j = 0; j < 8; ++j) {
                int kk = (j < 4) ? (8 * h + 2 * j) : (16 + 8 * h + 2 * (j - 4));
                afrag[mi][2 * j]     = ap[kk];
                afrag[mi][2 * j + 1] = ap[kk + 1];
            }
        }
#pragma unroll
        for (int ni = 0; ni < 4; ++ni) {
            v16bf bfrag;
            const __bf16* bp = &Bs[buf][(wn + ni * 16 + ml) * LDSP];
#pragma unroll
            for (int j = 0; j < 8; ++j) {
                int kk = (j < 4) ? (8 * h + 2 * j) : (16 + 8 * h + 2 * (j - 4));
                bfrag[2 * j]     = bp[kk];
                bfrag[2 * j + 1] = bp[kk + 1];
            }
#pragma unroll
            for (int mi = 0; mi < 2; ++mi) {
                acc[mi][ni] = __builtin_amdgcn_wmma_f32_16x16x32_bf16(
                    false, afrag[mi], false, bfrag, (short)0, acc[mi][ni],
                    false, false);
            }
        }
    }

    // --- epilogue: bias hoisted to 4 loads/lane; C/D layout:
    //     VGPR r -> M=r+8h, lane -> N=ml -----------------------------------
    float bb[4] = {0.0f, 0.0f, 0.0f, 0.0f};
    if (bias) {
#pragma unroll
        for (int ni = 0; ni < 4; ++ni)
            bb[ni] = bias[n0 + wn + ni * 16 + ml];
    }
#pragma unroll
    for (int mi = 0; mi < 2; ++mi)
#pragma unroll
        for (int ni = 0; ni < 4; ++ni)
#pragma unroll
            for (int r = 0; r < 8; ++r) {
                int mm = m0 + wm + mi * 16 + r + 8 * h;
                int nn = n0 + wn + ni * 16 + ml;
                float val = acc[mi][ni][r] * alpha + bb[ni];
                if (relu) val = fmaxf(val, 0.0f);
                C[(long)mm * ldc + nn] = (OT)val;
            }
}

// ---- row softmax: f32 scores -> bf16 probs --------------------------------
__global__ __launch_bounds__(256)
void softmax_rows(const float* __restrict__ S, __bf16* __restrict__ P, int T)
{
    __shared__ float red[8];
    const long row = blockIdx.x;
    const float* s = S + row * (long)T;
    __bf16*      p = P + row * (long)T;
    const int t = threadIdx.x, w = t >> 5;

    float mx = -3.4e38f;
    for (int i = t; i < T; i += 256) mx = fmaxf(mx, s[i]);
#pragma unroll
    for (int o = 16; o > 0; o >>= 1) mx = fmaxf(mx, __shfl_xor(mx, o, 32));
    if ((t & 31) == 0) red[w] = mx;
    __syncthreads();
    float rm = red[0];
#pragma unroll
    for (int i = 1; i < 8; ++i) rm = fmaxf(rm, red[i]);

    float sum = 0.0f;
    for (int i = t; i < T; i += 256) sum += __expf(s[i] - rm);
#pragma unroll
    for (int o = 16; o > 0; o >>= 1) sum += __shfl_xor(sum, o, 32);
    __syncthreads();
    if ((t & 31) == 0) red[w] = sum;
    __syncthreads();
    float tot = 0.0f;
#pragma unroll
    for (int i = 0; i < 8; ++i) tot += red[i];
    float inv = 1.0f / tot;

    for (int i = t; i < T; i += 256) p[i] = (__bf16)(__expf(s[i] - rm) * inv);
}

// ---- residual add + LayerNorm over D=512 (f32 math, bf16 io) --------------
__global__ __launch_bounds__(128)
void add_ln(const __bf16* __restrict__ X, const __bf16* __restrict__ Y,
            const float* __restrict__ g, const float* __restrict__ b,
            __bf16* __restrict__ Out)
{
    __shared__ float rs[4], rq[4];
    const long tok = blockIdx.x;
    const int t = threadIdx.x, w = t >> 5;
    const __bf16* x = X + tok * TD;
    const __bf16* y = Y + tok * TD;

    float v[4], sum = 0.0f, sq = 0.0f;
#pragma unroll
    for (int i = 0; i < 4; ++i) {
        int idx = t + i * 128;
        v[i] = (float)x[idx] + (float)y[idx];
        sum += v[i];
        sq  += v[i] * v[i];
    }
#pragma unroll
    for (int o = 16; o > 0; o >>= 1) {
        sum += __shfl_xor(sum, o, 32);
        sq  += __shfl_xor(sq,  o, 32);
    }
    if ((t & 31) == 0) { rs[w] = sum; rq[w] = sq; }
    __syncthreads();
    float ts = rs[0] + rs[1] + rs[2] + rs[3];
    float tq = rq[0] + rq[1] + rq[2] + rq[3];
    float mean = ts * (1.0f / TD);
    float var  = tq * (1.0f / TD) - mean * mean;
    float rstd = rsqrtf(var + 1e-5f);

    __bf16* o = Out + tok * TD;
#pragma unroll
    for (int i = 0; i < 4; ++i) {
        int idx = t + i * 128;
        o[idx] = (__bf16)((v[i] - mean) * rstd * g[idx] + b[idx]);
    }
}

// ---- sin positional-ish embedding: x = sin(ext * (d+1)) -------------------
__global__ void embed_kernel(const float* __restrict__ src,
                             const float* __restrict__ pred,
                             __bf16* __restrict__ X)
{
    long idx = (long)blockIdx.x * blockDim.x + threadIdx.x;  // over NTOK*TD
    int  d   = (int)(idx & (TD - 1));
    long tok = idx >> 9;
    int  e   = (int)(tok % TE);
    long bv  = tok / TE;
    float ext = (e < TL) ? src[bv * TL + e] : pred[bv];
    X[idx] = (__bf16)sinf(ext * (float)(d + 1));
}

// ---- (b,nv,e,d) -> (b,e,nv,d) transpose, 16B chunks -----------------------
__global__ void transpose_ev(const __bf16* __restrict__ in,
                             __bf16* __restrict__ out)
{
    long idx  = (long)blockIdx.x * blockDim.x + threadIdx.x; // NTOK*64 chunks
    int  c    = (int)(idx & 63);
    long orow = idx >> 6;                 // (b*TE+e)*TV + v
    int  v    = (int)(orow % TV);
    long be   = orow / TV;
    int  e    = (int)(be % TE);
    long b    = be / TE;
    long irow = (b * TV + v) * TE + e;
    ((uint4*)out)[orow * 64 + c] = ((const uint4*)in)[irow * 64 + c];
}

// ---- head: pred[b,v] (row e=95) @ lin_w^T + lin_b, f32 out ---------------
__global__ void head_kernel(const __bf16* __restrict__ X,
                            const float* __restrict__ lw,
                            const float* __restrict__ lb,
                            float* __restrict__ out)
{
    int idx = blockIdx.x * blockDim.x + threadIdx.x;
    if (idx >= TB * TV * TNC) return;
    int c  = idx % TNC;
    int bv = idx / TNC;
    int v  = bv % TV;
    int b  = bv / TV;
    long row = ((long)(b * TE + (TE - 1)) * TV + v);
    const __bf16* x = X + row * TD;
    const float*  w = lw + c * TD;
    float s = 0.0f;
    for (int k = 0; k < TD; ++k) s += (float)x[k] * w[k];
    out[idx] = s + lb[c];
}

// ---- f32 -> bf16 weight conversion ---------------------------------------
__global__ void f2bf(const float* __restrict__ in, __bf16* __restrict__ out,
                     long n)
{
    long i = (long)blockIdx.x * blockDim.x + threadIdx.x;
    if (i < n) out[i] = (__bf16)in[i];
}

// ---------------------------------------------------------------------------
extern "C" void kernel_launch(void* const* d_in, const int* in_sizes, int n_in,
                              void* d_out, int out_size, void* d_ws,
                              size_t ws_size, hipStream_t stream)
{
    (void)in_sizes; (void)n_in; (void)out_size; (void)ws_size;
    const float* src  = (const float*)d_in[0];
    // d_in[1] = masks (all-zero, unused by the reference)
    const float* pred = (const float*)d_in[2];
    const float* lin_w = (const float*)d_in[3];
    const float* lin_b = (const float*)d_in[4];
    const float* hp[12]; const float* vp[12];
    for (int i = 0; i < 12; ++i) hp[i] = (const float*)d_in[5 + i];
    for (int i = 0; i < 12; ++i) vp[i] = (const float*)d_in[17 + i];
    // param order: wqkv,bqkv,wo,bo,ln1g,ln1b,w1,b1,w2,b2,ln2g,ln2b

    // ---- workspace layout ----
    char* ws = (char*)d_ws;
    size_t off = 0;
    auto alloc = [&](size_t bytes) -> char* {
        off = (off + 255) & ~(size_t)255;
        char* p = ws + off;
        off += bytes;
        return p;
    };
    __bf16* wqkv_b[2]; __bf16* wo_b[2]; __bf16* w1_b[2]; __bf16* w2_b[2];
    for (int s = 0; s < 2; ++s) {
        wqkv_b[s] = (__bf16*)alloc((size_t)3 * TD * TD * 2);
        wo_b[s]   = (__bf16*)alloc((size_t)TD * TD * 2);
        w1_b[s]   = (__bf16*)alloc((size_t)TFF * TD * 2);
        w2_b[s]   = (__bf16*)alloc((size_t)TD * TFF * 2);
    }
    __bf16* x    = (__bf16*)alloc((size_t)NTOK * TD * 2);
    __bf16* y    = (__bf16*)alloc((size_t)NTOK * TD * 2);
    __bf16* zb   = (__bf16*)alloc((size_t)NTOK * TD * 2);
    __bf16* qkv  = (__bf16*)alloc((size_t)NTOK * 3 * TD * 2);
    float*  Sbuf = (float*)alloc((size_t)TV * (TB * TE) * (TB * TE) * 4); // 113MB (covers both stages)
    __bf16* Pbuf = (__bf16*)alloc((size_t)TV * (TB * TE) * (TB * TE) * 2);
    __bf16* hbuf = (__bf16*)alloc((size_t)NTOK * TFF * 2);

    // ---- convert weights to bf16 (stay resident in the 192MB L2) ----
    auto conv = [&](const float* in, __bf16* outp, long n) {
        f2bf<<<dim3((unsigned)((n + 255) / 256)), 256, 0, stream>>>(in, outp, n);
    };
    const float* const* pp[2] = { hp, vp };
    for (int s = 0; s < 2; ++s) {
        conv(pp[s][0], wqkv_b[s], (long)3 * TD * TD);
        conv(pp[s][2], wo_b[s],   (long)TD * TD);
        conv(pp[s][6], w1_b[s],   (long)TFF * TD);
        conv(pp[s][8], w2_b[s],   (long)TD * TFF);
    }

    // ---- embedding: x[(b*V+v)*E + e][d] = sin(ext * (d+1)) ----
    embed_kernel<<<dim3((unsigned)((long)NTOK * TD / 256)), 256, 0, stream>>>(
        src, pred, x);

    const float inv_sqrt_d = 0.04419417382415922f;  // 1/sqrt(512)

    auto run_layer = [&](int s, int Nslot, int Stot) {
        const int M = Nslot * Stot;  // 36864 in both stages
        const float* bqkv = pp[s][1];
        const float* bo   = pp[s][3];
        const float* ln1g = pp[s][4]; const float* ln1b = pp[s][5];
        const float* b1   = pp[s][7];
        const float* b2   = pp[s][9];
        const float* ln2g = pp[s][10]; const float* ln2b = pp[s][11];

        // 1) QKV projection: qkv = x @ Wqkv^T + bqkv        (M x 1536)
        gemm_wmma<__bf16, false><<<dim3(3 * TD / BN, M / BM, 1), 128, 0, stream>>>(
            x, wqkv_b[s], qkv, bqkv, M, 3 * TD, TD, TD, TD, 3 * TD,
            0, 0, 0, 0, 0, 0, 1.0f, 0);
        // 2) scores: S[n] = Q_n @ K_n^T / sqrt(D)           (per slot n)
        gemm_wmma<float, false><<<dim3(Stot / BN, Stot / BM, Nslot), 128, 0, stream>>>(
            qkv, qkv, Sbuf, nullptr, Stot, Stot, TD,
            (long)Nslot * 3 * TD, (long)Nslot * 3 * TD, Stot,
            0, 3 * TD, TD, 3 * TD, 0, (long)Stot * Stot, inv_sqrt_d, 0);
        // 3) softmax rows (L2-resident scores -> bf16 probs)
        softmax_rows<<<dim3((unsigned)(Nslot * Stot)), 256, 0, stream>>>(
            Sbuf, Pbuf, Stot);
        // 4) attn out: y[s,n] = P_n @ V_n
        gemm_wmma<__bf16, true><<<dim3(TD / BN, Stot / BM, Nslot), 128, 0, stream>>>(
            Pbuf, qkv, y, nullptr, Stot, TD, Stot,
            Stot, (long)Nslot * 3 * TD, (long)Nslot * TD,
            0, (long)Stot * Stot, 2 * TD, 3 * TD, 0, TD, 1.0f, 0);
        // 5) output projection: z = y @ Wo^T + bo
        gemm_wmma<__bf16, false><<<dim3(TD / BN, M / BM, 1), 128, 0, stream>>>(
            y, wo_b[s], zb, bo, M, TD, TD, TD, TD, TD,
            0, 0, 0, 0, 0, 0, 1.0f, 0);
        // 6) x = LN(x + z)
        add_ln<<<dim3((unsigned)M), 128, 0, stream>>>(x, zb, ln1g, ln1b, x);
        // 7) FFN1: h = relu(x @ W1^T + b1)
        gemm_wmma<__bf16, false><<<dim3(TFF / BN, M / BM, 1), 128, 0, stream>>>(
            x, w1_b[s], hbuf, b1, M, TFF, TD, TD, TD, TFF,
            0, 0, 0, 0, 0, 0, 1.0f, 1);
        // 8) FFN2: y = h @ W2^T + b2
        gemm_wmma<__bf16, false><<<dim3(TD / BN, M / BM, 1), 128, 0, stream>>>(
            hbuf, w2_b[s], y, b2, M, TD, TFF, TFF, TFF, TD,
            0, 0, 0, 0, 0, 0, 1.0f, 0);
        // 9) x = LN(x + y)
        add_ln<<<dim3((unsigned)M), 128, 0, stream>>>(x, y, ln2g, ln2b, x);
    };

    // stage 1: layout (b*V, E, D): attention over s-axis (384) per E-slot (96)
    for (int l = 0; l < 3; ++l) run_layer(0, TE, TB * TV);

    // transpose to (b*E, V, D)
    transpose_ev<<<dim3((unsigned)((long)NTOK * (TD / 8) / 256)), 256, 0, stream>>>(
        x, zb);
    { __bf16* t = x; x = zb; zb = t; }

    // stage 2: attention over (b*E)=768 per V-slot (48)
    for (int l = 0; l < 3; ++l) run_layer(1, TV, TB * TE);

    // head: rows with e == 95 -> (8,48,3) f32
    head_kernel<<<dim3((TB * TV * TNC + 255) / 256), 256, 0, stream>>>(
        x, lin_w, lin_b, (float*)d_out);
}
